// Dale_CB_STPcell_88691074662511
// MI455X (gfx1250) — compile-verified, matrix operand order checked by reading, not run on hardware
//
#include <hip/hip_runtime.h>
#include <hip/hip_bf16.h>

// ---------------------------------------------------------------------------
// Dale_CB_STP recurrent cell for MI455X (gfx1250), wave32 + WMMA bf16.
// H=512, IN=128, B=256, T=256, NC=10.
// ---------------------------------------------------------------------------

typedef __attribute__((ext_vector_type(16))) __bf16 v16bf;
typedef __attribute__((ext_vector_type(8)))  float  v8f;

union BF16x16 { v16bf v; uint4 q[2]; };

__device__ __forceinline__ float sigf(float x) {
    return 1.0f / (1.0f + __expf(-x));
}
__device__ __forceinline__ float softplusf(float x) {
    return fmaxf(x, 0.0f) + log1pf(__expf(-fabsf(x)));
}
// pack two floats into one dword of two bf16 (RNE via scalar cast)
__device__ __forceinline__ unsigned int pack2bf(float a, float b) {
    union { __bf16 h; unsigned short s; } ua, ub;
    ua.h = (__bf16)a; ub.h = (__bf16)b;
    return (unsigned int)ua.s | ((unsigned int)ub.s << 16);
}

#define H_DIM 512
#define IN_DIM 128
#define B_DIM 256
#define T_DIM 256
#define NC_DIM 10

// ---------------------------------------------------------------------------
// Kernel 1: one-time precompute. softplus(K), Dale-combined W, bf16 casts of
// P / P_z, and the sigmoid-derived per-row constants z_x, z_u, Ucap.
// ---------------------------------------------------------------------------
__global__ void prep_kernel(const float* __restrict__ K, const float* __restrict__ C,
                            const float* __restrict__ P, const float* __restrict__ Pz,
                            const float* __restrict__ c_x, const float* __restrict__ c_u,
                            const float* __restrict__ c_U,
                            const float* __restrict__ e_e, const float* __restrict__ e_i,
                            __bf16* __restrict__ Kbf, __bf16* __restrict__ Wbf,
                            __bf16* __restrict__ Pbf, __bf16* __restrict__ Pzbf,
                            float* __restrict__ zx, float* __restrict__ zu,
                            float* __restrict__ Uc) {
    int idx = blockIdx.x * blockDim.x + threadIdx.x;
    float ee = e_e[0], ei = e_i[0];
    if (idx < H_DIM * H_DIM) {
        float ksp = softplusf(K[idx]);
        float csp = softplusf(C[idx]);
        Kbf[idx] = (__bf16)ksp;
        float s = ksp + csp;
        int col = idx & (H_DIM - 1);
        float w = (col < H_DIM / 2) ? fmaxf(ee * s, 0.0f) : -fmaxf(-(ei * s), 0.0f);
        Wbf[idx] = (__bf16)w;
    }
    if (idx < H_DIM * IN_DIM) {
        Pbf[idx]  = (__bf16)P[idx];
        Pzbf[idx] = (__bf16)Pz[idx];
    }
    if (idx < H_DIM) {
        zx[idx] = 0.001f + 0.099f * sigf(c_x[idx]);
        zu[idx] = 0.001f + 0.099f * sigf(c_u[idx]);
        Uc[idx] = 0.9f * sigf(c_U[idx]);
    }
}

// ---------------------------------------------------------------------------
// Kernel 2: persistent recurrent scan. 16 blocks (one 16-wide batch tile each),
// 512 threads = 16 waves; each wave owns two 16-row output tiles. State lives
// in registers in the WMMA C-fragment layout; r and s are shared via LDS in a
// transposed [col][k] layout so B fragments are contiguous ds_load_b128s.
// A (weight) fragments are streamed from L2 every step; an asm memory clobber
// per iteration stops LICM from hoisting+spilling them to scratch.
// ---------------------------------------------------------------------------
__global__ __launch_bounds__(512, 1) void rnn_kernel(
    const float*  __restrict__ x,
    const __bf16* __restrict__ Ksp, const __bf16* __restrict__ W,
    const __bf16* __restrict__ Pm,  const __bf16* __restrict__ Pz,
    const float*  __restrict__ zxg, const float* __restrict__ zug,
    const float*  __restrict__ Ucg,
    const float*  __restrict__ bz,  const float* __restrict__ bv,
    float* __restrict__ vstore)
{
    // padded rows: 528 * 2B = 1056B stride (16B multiple), spreads LDS banks
    __shared__ __align__(16) __bf16 rT[16 * 528];
    __shared__ __align__(16) __bf16 sT[16 * 528];
    __shared__ __align__(16) __bf16 xT[16 * 144];

    const int tid  = threadIdx.x;
    const int w    = tid >> 5;       // wave id 0..15
    const int lane = tid & 31;
    const int g    = lane >> 4;      // lane group (ISA A/B fragment geometry)
    const int n    = lane & 15;      // row-in-tile (A) / batch col (B,C,D)
    const int bcol = blockIdx.x * 16 + n;

    const int rowb0 = (w * 2 + 0) * 16;
    const int rowb1 = (w * 2 + 1) * 16;

    v8f zxv[2], zuv[2], Ucv[2], bzv[2], bvv[2];
    v8f vS[2], Xs[2], Us[2];
#pragma unroll
    for (int i = 0; i < 2; ++i) {
        int rb = i ? rowb1 : rowb0;
#pragma unroll
        for (int j = 0; j < 8; ++j) {
            int row = rb + j + 8 * g;          // C-fragment row for element j
            zxv[i][j] = zxg[row];
            zuv[i][j] = zug[row];
            Ucv[i][j] = Ucg[row];
            bzv[i][j] = bz[row];
            bvv[i][j] = bv[row];
            vS[i][j] = 0.0f;
            Xs[i][j] = 1.0f;
            Us[i][j] = Ucv[i][j];              // U0 = Ucap
        }
    }

    // per-lane A-fragment element offsets (lane-group K offset folded in)
    const int aoffH0 = (rowb0 + n) * H_DIM  + g * 8;   // Ksp / W row, tile 0
    const int aoffH1 = (rowb1 + n) * H_DIM  + g * 8;   // Ksp / W row, tile 1
    const int aoffI0 = (rowb0 + n) * IN_DIM + g * 8;   // P / P_z row, tile 0
    const int aoffI1 = (rowb1 + n) * IN_DIM + g * 8;   // P / P_z row, tile 1
    const int rTl = n * 528 + g * 8;   // B-fragment lane base into rT / sT
    const int xTl = n * 144 + g * 8;   // B-fragment lane base into xT

    for (int t = 0; t < T_DIM; ++t) {
        __syncthreads();   // previous step's GEMM reads done before overwrite
        // stop LICM from pre-loading loop-invariant weights (spill disaster)
        asm volatile("" ::: "memory");

        // ---- stage x_t (f32 -> bf16) into LDS, [col][k] = already K-major ----
#pragma unroll
        for (int it = 0; it < 4; ++it) {
            int idx = tid + it * 512;
            int col = idx >> 7, k = idx & 127;
            float xv = x[((size_t)(blockIdx.x * 16 + col) * T_DIM + t) * IN_DIM + k];
            xT[col * 144 + k] = (__bf16)xv;
        }

        // ---- elementwise STP update; emit r and s = Un*Xn*r to LDS ----
#pragma unroll
        for (int i = 0; i < 2; ++i) {
            int rb = i ? rowb1 : rowb0;
            float rv[8], sv[8];
#pragma unroll
            for (int j = 0; j < 8; ++j) {
                float r  = sigf(vS[i][j]);                 // uses pre-update v
                float Xo = Xs[i][j], Uo = Us[i][j];
                float zx = zxv[i][j], zu = zuv[i][j], Uc = Ucv[i][j];
                float Xn = zx + (1.0f - zx) * Xo - Uo * Xo * r;
                float Un = Uc * zu + (1.0f - zu) * Uo + Uc * (1.0f - Uo) * r;
                Un = fminf(fmaxf(Un, Uc), 1.0f);
                Xs[i][j] = Xn;
                Us[i][j] = Un;
                rv[j] = r;
                sv[j] = Un * Xn * r;
            }
            uint4 rq, sq;
            rq.x = pack2bf(rv[0], rv[1]); rq.y = pack2bf(rv[2], rv[3]);
            rq.z = pack2bf(rv[4], rv[5]); rq.w = pack2bf(rv[6], rv[7]);
            sq.x = pack2bf(sv[0], sv[1]); sq.y = pack2bf(sv[2], sv[3]);
            sq.z = pack2bf(sv[4], sv[5]); sq.w = pack2bf(sv[6], sv[7]);
            int off = n * 528 + rb + 8 * g;   // 8 consecutive rows, 16B store
            *(uint4*)&rT[off] = rq;
            *(uint4*)&sT[off] = sq;
        }
        __syncthreads();

        // ---- z_pre = Ksp @ r + P_z @ x_t + b_z ----
        v8f accz0 = bzv[0], accz1 = bzv[1];
#pragma unroll
        for (int kc = 0; kc < 16; ++kc) {
            BF16x16 bfr, a0, a1;
            const __bf16* bp = &rT[rTl + kc * 32];
            bfr.q[0] = *(const uint4*)bp;
            bfr.q[1] = *(const uint4*)(bp + 16);
            const __bf16* ap0 = Ksp + aoffH0 + kc * 32;
            const __bf16* ap1 = Ksp + aoffH1 + kc * 32;
            a0.q[0] = *(const uint4*)ap0; a0.q[1] = *(const uint4*)(ap0 + 16);
            a1.q[0] = *(const uint4*)ap1; a1.q[1] = *(const uint4*)(ap1 + 16);
            accz0 = __builtin_amdgcn_wmma_f32_16x16x32_bf16(false, a0.v, false, bfr.v, (short)0, accz0, false, false);
            accz1 = __builtin_amdgcn_wmma_f32_16x16x32_bf16(false, a1.v, false, bfr.v, (short)0, accz1, false, false);
        }
#pragma unroll
        for (int kc = 0; kc < 4; ++kc) {
            BF16x16 bfr, a0, a1;
            const __bf16* bp = &xT[xTl + kc * 32];
            bfr.q[0] = *(const uint4*)bp;
            bfr.q[1] = *(const uint4*)(bp + 16);
            const __bf16* ap0 = Pz + aoffI0 + kc * 32;
            const __bf16* ap1 = Pz + aoffI1 + kc * 32;
            a0.q[0] = *(const uint4*)ap0; a0.q[1] = *(const uint4*)(ap0 + 16);
            a1.q[0] = *(const uint4*)ap1; a1.q[1] = *(const uint4*)(ap1 + 16);
            accz0 = __builtin_amdgcn_wmma_f32_16x16x32_bf16(false, a0.v, false, bfr.v, (short)0, accz0, false, false);
            accz1 = __builtin_amdgcn_wmma_f32_16x16x32_bf16(false, a1.v, false, bfr.v, (short)0, accz1, false, false);
        }

        // ---- v_pre = W @ s + P @ x_t + b_v ----
        v8f accv0 = bvv[0], accv1 = bvv[1];
#pragma unroll
        for (int kc = 0; kc < 16; ++kc) {
            BF16x16 bfr, a0, a1;
            const __bf16* bp = &sT[rTl + kc * 32];
            bfr.q[0] = *(const uint4*)bp;
            bfr.q[1] = *(const uint4*)(bp + 16);
            const __bf16* ap0 = W + aoffH0 + kc * 32;
            const __bf16* ap1 = W + aoffH1 + kc * 32;
            a0.q[0] = *(const uint4*)ap0; a0.q[1] = *(const uint4*)(ap0 + 16);
            a1.q[0] = *(const uint4*)ap1; a1.q[1] = *(const uint4*)(ap1 + 16);
            accv0 = __builtin_amdgcn_wmma_f32_16x16x32_bf16(false, a0.v, false, bfr.v, (short)0, accv0, false, false);
            accv1 = __builtin_amdgcn_wmma_f32_16x16x32_bf16(false, a1.v, false, bfr.v, (short)0, accv1, false, false);
        }
#pragma unroll
        for (int kc = 0; kc < 4; ++kc) {
            BF16x16 bfr, a0, a1;
            const __bf16* bp = &xT[xTl + kc * 32];
            bfr.q[0] = *(const uint4*)bp;
            bfr.q[1] = *(const uint4*)(bp + 16);
            const __bf16* ap0 = Pm + aoffI0 + kc * 32;
            const __bf16* ap1 = Pm + aoffI1 + kc * 32;
            a0.q[0] = *(const uint4*)ap0; a0.q[1] = *(const uint4*)(ap0 + 16);
            a1.q[0] = *(const uint4*)ap1; a1.q[1] = *(const uint4*)(ap1 + 16);
            accv0 = __builtin_amdgcn_wmma_f32_16x16x32_bf16(false, a0.v, false, bfr.v, (short)0, accv0, false, false);
            accv1 = __builtin_amdgcn_wmma_f32_16x16x32_bf16(false, a1.v, false, bfr.v, (short)0, accv1, false, false);
        }

        // ---- v <- (1 - DT*sigmoid(z_pre)) * v + DT * v_pre ----
#pragma unroll
        for (int j = 0; j < 8; ++j) {
            float zt0 = 0.1f * sigf(accz0[j]);
            vS[0][j] = (1.0f - zt0) * vS[0][j] + 0.1f * accv0[j];
            float zt1 = 0.1f * sigf(accz1[j]);
            vS[1][j] = (1.0f - zt1) * vS[1][j] + 0.1f * accv1[j];
        }
    }

    // ---- dump excitatory half of v_f (rows < 256) ----
#pragma unroll
    for (int i = 0; i < 2; ++i) {
        int rb = i ? rowb1 : rowb0;
        if (rb < H_DIM / 2) {
#pragma unroll
            for (int j = 0; j < 8; ++j) {
                int row = rb + j + 8 * g;
                vstore[row * B_DIM + bcol] = vS[i][j];
            }
        }
    }
}

// ---------------------------------------------------------------------------
// Kernel 3: out[b, c] = fc_b[c] + sum_{h<256} vf[h, b] * fc_w[c, h]
// ---------------------------------------------------------------------------
__global__ void fc_kernel(const float* __restrict__ vstore,
                          const float* __restrict__ fc_w,
                          const float* __restrict__ fc_b,
                          float* __restrict__ out) {
    int c = blockIdx.x;     // 0..9
    int b = threadIdx.x;    // 0..255
    float acc = fc_b[c];
    for (int h = 0; h < H_DIM / 2; ++h)
        acc = fmaf(vstore[h * B_DIM + b], fc_w[c * H_DIM + h], acc);
    out[b * NC_DIM + c] = acc;
}

// ---------------------------------------------------------------------------
extern "C" void kernel_launch(void* const* d_in, const int* in_sizes, int n_in,
                              void* d_out, int out_size, void* d_ws, size_t ws_size,
                              hipStream_t stream) {
    (void)in_sizes; (void)n_in; (void)out_size; (void)ws_size;

    const float* x    = (const float*)d_in[0];
    const float* P    = (const float*)d_in[1];
    const float* b_v  = (const float*)d_in[2];
    const float* K    = (const float*)d_in[3];
    const float* C    = (const float*)d_in[4];
    const float* P_z  = (const float*)d_in[5];
    const float* b_z  = (const float*)d_in[6];
    const float* e_e  = (const float*)d_in[7];
    const float* e_i  = (const float*)d_in[8];
    const float* c_x  = (const float*)d_in[9];
    const float* c_u  = (const float*)d_in[10];
    const float* c_U  = (const float*)d_in[11];
    const float* fc_w = (const float*)d_in[12];
    const float* fc_b = (const float*)d_in[13];

    char* ws = (char*)d_ws;
    __bf16* Kbf  = (__bf16*)(ws + 0);          // 512*512*2 = 524288
    __bf16* Wbf  = (__bf16*)(ws + 524288);     // 524288
    __bf16* Pbf  = (__bf16*)(ws + 1048576);    // 512*128*2 = 131072
    __bf16* Pzbf = (__bf16*)(ws + 1179648);    // 131072
    float*  zx   = (float*)(ws + 1310720);     // 2048
    float*  zu   = (float*)(ws + 1312768);     // 2048
    float*  Uc   = (float*)(ws + 1314816);     // 2048
    float*  vst  = (float*)(ws + 1316864);     // 256*256*4 = 262144

    prep_kernel<<<1024, 256, 0, stream>>>(K, C, P, P_z, c_x, c_u, c_U, e_e, e_i,
                                          Kbf, Wbf, Pbf, Pzbf, zx, zu, Uc);
    rnn_kernel<<<16, 512, 0, stream>>>(x, Kbf, Wbf, Pbf, Pzbf, zx, zu, Uc,
                                       b_z, b_v, vst);
    fc_kernel<<<NC_DIM, B_DIM, 0, stream>>>(vst, fc_w, fc_b, (float*)d_out);
}
